// GeneratorSDE_12025908428803
// MI455X (gfx1250) — compile-verified, matrix operand order checked by reading, not run on hardware
//
#include <hip/hip_runtime.h>
#include <math.h>

typedef __attribute__((ext_vector_type(16))) _Float16 v16h;
typedef __attribute__((ext_vector_type(8)))  _Float16 v8h;
typedef __attribute__((ext_vector_type(8)))  float    v8f;

#define N_PATHS 8192
#define SEQ_LEN 512
#define BM 64          // paths per block
#define NT 128         // threads per block (4 wave32)
#define MIN_VOL 1e-4f
#define MAX_VOL 5.0f

__device__ __forceinline__ v16h ldfrag(const _Float16* p) {
  // A/B fragment: VGPR0-3 = K[base..base+7], VGPR4-7 = K[base+16..base+23]
  v16h r;
  v8h lo = *(const v8h*)p;          // 16B aligned -> ds_load_b128
  v8h hi = *(const v8h*)(p + 16);
#pragma unroll
  for (int i = 0; i < 8; ++i) { r[i] = lo[i]; r[8 + i] = hi[i]; }
  return r;
}

// fast sigmoid: v_exp_f32 + v_rcp_f32 (no IEEE divide expansion)
__device__ __forceinline__ float sigmoidf_(float x) {
  return __builtin_amdgcn_rcpf(1.f + __expf(-x));
}
// fast softplus: v_exp_f32 + v_log_f32
__device__ __forceinline__ float softplusf_(float x) {
  return x > 20.f ? x : __logf(1.f + __expf(x));
}

__global__ __launch_bounds__(NT) void sde_milstein_kernel(
    const float* __restrict__ y0g, const float* __restrict__ tsg,
    const float* __restrict__ dWg,
    const float* __restrict__ fw1g, const float* __restrict__ fb1g,
    const float* __restrict__ fw2g, const float* __restrict__ fb2g,
    const float* __restrict__ fw3g, const float* __restrict__ fb3g,
    const float* __restrict__ gw1g, const float* __restrict__ gb1g,
    const float* __restrict__ gw2g, const float* __restrict__ gb2g,
    const float* __restrict__ gw3g, const float* __restrict__ gb3g,
    float* __restrict__ outp)
{
  // ---- weights (f16, BT layout: [n][k], k-contiguous, stride 136 = 16B-aligned rows)
  __shared__ __align__(16) _Float16 wfw2t[128][136];  // fw2^T  : [n][k] = fw2[k][n]
  __shared__ __align__(16) _Float16 wgw2t[128][136];  // gw2^T  : [n][k] = gw2[k][n]
  __shared__ __align__(16) _Float16 wgw2r[128][136];  // gw2 row: [n][k] = gw2[n][k]  (B of backward GEMM)
  __shared__ __align__(16) _Float16 fw1p[128][40];    // fw1 padded K: [n][k<3 valid]
  __shared__ __align__(16) _Float16 gw1p[128][40];    // gw1 padded K: [n][k<2 valid]
  __shared__ __align__(16) _Float16 fw3p[16][136];    // fw3 head: [n<2][k]
  __shared__ __align__(16) _Float16 gw3p[16][136];    // gw3 head
  __shared__ __align__(16) _Float16 gw1tp[16][136];   // gw1 (corr head): [n<2][k] = gw1[n][k]
  // ---- activations
  __shared__ __align__(16) _Float16 yA[BM][40];       // [y0,y1,spread,0...] A-matrix, K padded to 32
  __shared__ __align__(16) _Float16 hf[BM][136];      // drift: h1 then h2
  __shared__ __align__(16) _Float16 af[BM][136];      // diff : a1 -> a2 -> dz2 -> dz1
  __shared__ __align__(16) _Float16 sp1s[BM][128];    // silu'(z1_diff)
  __shared__ __align__(16) _Float16 sp2s[BM][128];    // silu'(z2_diff)
  // ---- small f32
  __shared__ float fb1s[128], fb2s[128], gb1s[128], gb2s[128];
  __shared__ float fb3s[2], gb3s[2];
  __shared__ float gw3f[128][2];
  __shared__ float ys[BM][2], fs[BM][2], gs[BM][2], us[BM][2], corrs[BM][2], dwc[BM][2];

  const int tid  = threadIdx.x;
  const int L    = tid & 31;      // lane in wave32
  const int wv   = tid >> 5;      // wave id 0..3
  const int ln   = L & 15;
  const int half = L >> 4;
  const int kb8  = half * 8;      // K chunk base for this lane half
  const int arow = wv * 16 + ln;  // A-matrix row this lane owns
  const int blockBase = blockIdx.x * BM;

  // ================= one-time staging =================
  for (int i = tid; i < 128 * 128; i += NT) {
    int r = i >> 7, c = i & 127;                 // element [r][c]
    float f2 = fw2g[i], g2 = gw2g[i];
    wfw2t[c][r] = (_Float16)f2;
    wgw2t[c][r] = (_Float16)g2;
    wgw2r[r][c] = (_Float16)g2;
  }
  for (int i = tid; i < 128 * 40; i += NT) {
    int n = i / 40, k = i % 40;
    fw1p[n][k] = (_Float16)((k < 3) ? fw1g[k * 128 + n] : 0.f);
    gw1p[n][k] = (_Float16)((k < 2) ? gw1g[k * 128 + n] : 0.f);
  }
  for (int i = tid; i < 16 * 136; i += NT) {
    int n = i / 136, k = i % 136;
    bool ok = (n < 2) && (k < 128);
    fw3p[n][k]  = (_Float16)(ok ? fw3g[k * 2 + n] : 0.f);
    gw3p[n][k]  = (_Float16)(ok ? gw3g[k * 2 + n] : 0.f);
    gw1tp[n][k] = (_Float16)(ok ? gw1g[n * 128 + k] : 0.f);
  }
  {
    fb1s[tid] = fb1g[tid]; fb2s[tid] = fb2g[tid];
    gb1s[tid] = gb1g[tid]; gb2s[tid] = gb2g[tid];
  }
  if (tid < 2) { fb3s[tid] = fb3g[tid]; gb3s[tid] = gb3g[tid]; }
  for (int i = tid; i < 256; i += NT) gw3f[i >> 1][i & 1] = gw3g[i];
  for (int i = tid; i < BM * 40; i += NT) yA[i / 40][i % 40] = (_Float16)0.f;
  {
    int m = tid >> 1, d = tid & 1;
    float y = y0g[(blockBase + m) * 2 + d];
    ys[m][d] = y;
    outp[(size_t)(blockBase + m) * (SEQ_LEN * 2) + d] = y;   // t = 0
  }
  __syncthreads();

  // ================= time loop =================
#pragma unroll 1
  for (int t = 0; t < SEQ_LEN - 1; ++t) {
    float dt = tsg[t + 1] - tsg[t];
    float sdt = sqrtf(dt);
    // issue the HBM load now, consume (store to LDS) only before the heads
    // stage -> latency hidden under the L1/L2 WMMA work.
    const int um = tid >> 1, ud = tid & 1;
    float dwreg = dWg[((size_t)t * N_PATHS + blockBase + um) * 2 + ud] * sdt;

    if (tid < BM) {
      float a = ys[tid][0], b = ys[tid][1];
      yA[tid][0] = (_Float16)a; yA[tid][1] = (_Float16)b; yA[tid][2] = (_Float16)(a - b);
    }
    __syncthreads();

    // ---- layer 1 (K padded to 32): drift -> hf, diffusion -> af (+ sp1s)
    {
      v16h a1 = ldfrag(&yA[arow][kb8]);
#pragma unroll
      for (int nt = 0; nt < 8; ++nt) {
        int col = nt * 16 + ln;
        v16h bf = ldfrag(&fw1p[col][kb8]);
        v8f c = {};
        c = __builtin_amdgcn_wmma_f32_16x16x32_f16(false, a1, false, bf, (short)0, c, false, false);
        float bias = fb1s[col];
#pragma unroll
        for (int r = 0; r < 8; ++r) {
          int row = wv * 16 + half * 8 + r;
          float z = c[r] + bias, s = sigmoidf_(z);
          hf[row][col] = (_Float16)(z * s);
        }
        v16h bg = ldfrag(&gw1p[col][kb8]);
        v8f cg = {};
        cg = __builtin_amdgcn_wmma_f32_16x16x32_f16(false, a1, false, bg, (short)0, cg, false, false);
        float biasg = gb1s[col];
#pragma unroll
        for (int r = 0; r < 8; ++r) {
          int row = wv * 16 + half * 8 + r;
          float z = cg[r] + biasg, s = sigmoidf_(z);
          float sil = z * s;
          af[row][col]   = (_Float16)sil;
          sp1s[row][col] = (_Float16)__builtin_fmaf(sil, 1.f - s, s);  // silu'
        }
      }
    }

    // ---- layer 2 (K=128): drift h2 -> hf, diffusion a2 -> af (+ sp2s)
    {
      v16h a[4], ag[4];
#pragma unroll
      for (int kt = 0; kt < 4; ++kt) a[kt]  = ldfrag(&hf[arow][kt * 32 + kb8]);
#pragma unroll
      for (int kt = 0; kt < 4; ++kt) ag[kt] = ldfrag(&af[arow][kt * 32 + kb8]);
#pragma unroll
      for (int nt = 0; nt < 8; ++nt) {
        int col = nt * 16 + ln;
        v8f c = {};
#pragma unroll
        for (int kt = 0; kt < 4; ++kt) {
          v16h b = ldfrag(&wfw2t[col][kt * 32 + kb8]);
          c = __builtin_amdgcn_wmma_f32_16x16x32_f16(false, a[kt], false, b, (short)0, c, false, false);
        }
        float bias = fb2s[col];
#pragma unroll
        for (int r = 0; r < 8; ++r) {
          int row = wv * 16 + half * 8 + r;
          float z = c[r] + bias, s = sigmoidf_(z);
          hf[row][col] = (_Float16)(z * s);
        }
        v8f cg = {};
#pragma unroll
        for (int kt = 0; kt < 4; ++kt) {
          v16h b = ldfrag(&wgw2t[col][kt * 32 + kb8]);
          cg = __builtin_amdgcn_wmma_f32_16x16x32_f16(false, ag[kt], false, b, (short)0, cg, false, false);
        }
        float biasg = gb2s[col];
#pragma unroll
        for (int r = 0; r < 8; ++r) {
          int row = wv * 16 + half * 8 + r;
          float z = cg[r] + biasg, s = sigmoidf_(z);
          float sil = z * s;
          af[row][col]   = (_Float16)sil;
          sp2s[row][col] = (_Float16)__builtin_fmaf(sil, 1.f - s, s);
        }
      }
    }

    // park dW into LDS now (same-wave consumers in the heads stage)
    dwc[um][ud] = dwreg;

    // ---- output heads (N padded to 16): f = h2@fw3+fb3 ; raw = a2@gw3+gb3 -> g, u
    {
      v16h a[4], ag[4];
#pragma unroll
      for (int kt = 0; kt < 4; ++kt) a[kt]  = ldfrag(&hf[arow][kt * 32 + kb8]);
#pragma unroll
      for (int kt = 0; kt < 4; ++kt) ag[kt] = ldfrag(&af[arow][kt * 32 + kb8]);
      v8f c = {}, cg = {};
#pragma unroll
      for (int kt = 0; kt < 4; ++kt) {
        v16h b = ldfrag(&fw3p[ln][kt * 32 + kb8]);
        c = __builtin_amdgcn_wmma_f32_16x16x32_f16(false, a[kt], false, b, (short)0, c, false, false);
      }
#pragma unroll
      for (int kt = 0; kt < 4; ++kt) {
        v16h b = ldfrag(&gw3p[ln][kt * 32 + kb8]);
        cg = __builtin_amdgcn_wmma_f32_16x16x32_f16(false, ag[kt], false, b, (short)0, cg, false, false);
      }
      if (ln < 2) {
        float bias = fb3s[ln], biasg = gb3s[ln];
#pragma unroll
        for (int r = 0; r < 8; ++r) {
          int row = wv * 16 + half * 8 + r;
          fs[row][ln] = c[r] + bias;
          float raw = cg[r] + biasg;
          float sp  = softplusf_(raw);
          float g   = fminf(fmaxf(sp, MIN_VOL), MAX_VOL);
          gs[row][ln] = g;
          float dw = dwc[row][ln];
          float v  = 0.5f * (dw * dw - dt);
          float mask = (sp >= MIN_VOL && sp <= MAX_VOL) ? 1.f : 0.f;
          us[row][ln] = v * g * mask * sigmoidf_(raw);   // cotangent into raw
        }
      }
    }
    __syncthreads();

    // ---- VJP stage 1: dz2 = (u @ gw3^T) * silu'(z2)  -> af (f16 A-matrix)
    {
      float w0 = gw3f[tid][0], w1 = gw3f[tid][1];   // column j = tid is fixed per thread
#pragma unroll 4
      for (int m = 0; m < BM; ++m) {
        float dh2 = us[m][0] * w0 + us[m][1] * w1;
        af[m][tid] = (_Float16)(dh2 * (float)sp2s[m][tid]);
      }
    }
    __syncthreads();

    // ---- VJP stage 2 (K=128): dh1 = dz2 @ gw2^T ; dz1 = dh1 * silu'(z1) -> af
    {
      v16h a[4];
#pragma unroll
      for (int kt = 0; kt < 4; ++kt) a[kt] = ldfrag(&af[arow][kt * 32 + kb8]);
#pragma unroll
      for (int nt = 0; nt < 8; ++nt) {
        int col = nt * 16 + ln;
        v8f c = {};
#pragma unroll
        for (int kt = 0; kt < 4; ++kt) {
          v16h b = ldfrag(&wgw2r[col][kt * 32 + kb8]);
          c = __builtin_amdgcn_wmma_f32_16x16x32_f16(false, a[kt], false, b, (short)0, c, false, false);
        }
#pragma unroll
        for (int r = 0; r < 8; ++r) {
          int row = wv * 16 + half * 8 + r;
          af[row][col] = (_Float16)(c[r] * (float)sp1s[row][col]);
        }
      }
    }
    // ---- VJP stage 3: corr = dz1 @ gw1^T (N padded to 16)
    {
      v16h a[4];
#pragma unroll
      for (int kt = 0; kt < 4; ++kt) a[kt] = ldfrag(&af[arow][kt * 32 + kb8]);
      v8f c = {};
#pragma unroll
      for (int kt = 0; kt < 4; ++kt) {
        v16h b = ldfrag(&gw1tp[ln][kt * 32 + kb8]);
        c = __builtin_amdgcn_wmma_f32_16x16x32_f16(false, a[kt], false, b, (short)0, c, false, false);
      }
      if (ln < 2) {
#pragma unroll
        for (int r = 0; r < 8; ++r) {
          int row = wv * 16 + half * 8 + r;
          corrs[row][ln] = c[r];
        }
      }
    }
    // ---- state update (all producer data is same-wave)
    {
      float y = ys[um][ud] + fs[um][ud] * dt + gs[um][ud] * dwc[um][ud] + corrs[um][ud];
      ys[um][ud] = y;
      outp[(size_t)(blockBase + um) * (SEQ_LEN * 2) + (t + 1) * 2 + ud] = y;
    }
    __syncthreads();
  }
}

extern "C" void kernel_launch(void* const* d_in, const int* in_sizes, int n_in,
                              void* d_out, int out_size, void* d_ws, size_t ws_size,
                              hipStream_t stream) {
  const float* y0  = (const float*)d_in[0];
  const float* ts  = (const float*)d_in[1];
  const float* dW  = (const float*)d_in[2];
  const float* fw1 = (const float*)d_in[3];
  const float* fb1 = (const float*)d_in[4];
  const float* fw2 = (const float*)d_in[5];
  const float* fb2 = (const float*)d_in[6];
  const float* fw3 = (const float*)d_in[7];
  const float* fb3 = (const float*)d_in[8];
  const float* gw1 = (const float*)d_in[9];
  const float* gb1 = (const float*)d_in[10];
  const float* gw2 = (const float*)d_in[11];
  const float* gb2 = (const float*)d_in[12];
  const float* gw3 = (const float*)d_in[13];
  const float* gb3 = (const float*)d_in[14];
  float* out = (float*)d_out;

  dim3 grid(N_PATHS / BM), block(NT);
  hipLaunchKernelGGL(sde_milstein_kernel, grid, block, 0, stream,
                     y0, ts, dW, fw1, fb1, fw2, fb2, fw3, fb3,
                     gw1, gb1, gw2, gb2, gw3, gb3, out);
}
